// AutoCorrelation_22007412424957
// MI455X (gfx1250) — matrix operand order repaired and checked by reading, not compiled, not run on hardware
//
#include <hip/hip_runtime.h>

typedef __attribute__((ext_vector_type(8)))  __bf16 v8bf;
typedef __attribute__((ext_vector_type(16))) __bf16 v16bf;
typedef __attribute__((ext_vector_type(8)))  float  v8f;
typedef __attribute__((ext_vector_type(4)))  float  v4f;

#define TT        4096
#define COPYBLKS  768        // 768 blocks of 8 bf16 = 6144 elements per shifted copy
#define COPYLEN   (COPYBLKS * 8)
#define XROWLEN   6160       // ring-extended fp32 row (max index used: 6151)
#define RSIZE     2113       // 2049 lags + stride padding (idx + idx>>5)
#define NTHREADS  512        // 16 waves of 32

__device__ __forceinline__ __bf16 f2bf(float f) {
  // round-to-nearest-even fp32 -> bf16
  unsigned u = __builtin_bit_cast(unsigned, f);
  u += 0x7FFFu + ((u >> 16) & 1u);
  unsigned short h = (unsigned short)(u >> 16);
  return __builtin_bit_cast(__bf16, h);
}

__device__ __forceinline__ v16bf cat16(v8bf lo, v8bf hi) {
  return __builtin_shufflevector(lo, hi, 0,1,2,3,4,5,6,7,8,9,10,11,12,13,14,15);
}

// 16B-block XOR swizzle: kills the 256B-periodic lane stride (128 bf16) of B fetches
__device__ __forceinline__ int swz(int blk) { return blk ^ ((blk >> 4) & 15); }

// stride-padded index into the lag accumulator array (banks spread for ds_add_f32)
__device__ __forceinline__ int ridx(int lag) { return lag + (lag >> 5); }

__global__ __launch_bounds__(NTHREADS, 1)
void AutoCorrelation_gfx1250_kernel(const float* __restrict__ x,
                                    float* __restrict__ out) {
  // LDS: 96KB swizzled bf16 copies + 24KB fp32 ring-extended row + 8.3KB lags + 4KB reduce
  __shared__ __align__(16) __bf16 xcop[8][COPYLEN];
  __shared__ __align__(16) float  xrowext[XROWLEN];
  __shared__ float rarr[RSIZE];
  __shared__ float rv[NTHREADS];
  __shared__ int   ri[NTHREADS];
  __shared__ int   lagsSh[3];

  const int tid = threadIdx.x;
  const long long base = (long long)blockIdx.x * TT;

  // ---- Stage 0: ring-extended fp32 row from HBM; zero lag accumulators ----
  for (int i = tid; i < XROWLEN; i += NTHREADS) xrowext[i] = x[base + (i & (TT - 1))];
  for (int i = tid; i < RSIZE; i += NTHREADS) rarr[i] = 0.0f;
  __syncthreads();

  // ---- Stage 1 (vectorized): 8 element-shifted bf16 copies, 16B-block swizzled ----
  // logical xcop[p][i] = bf16(x[(i+p) mod T]); physical block = swz(i>>3)
  for (int blk = tid; blk < COPYBLKS; blk += NTHREADS) {
    const float* src = &xrowext[blk * 8];
    v4f w0 = *(const v4f*)(src + 0);
    v4f w1 = *(const v4f*)(src + 4);
    v4f w2 = *(const v4f*)(src + 8);
    v4f w3 = *(const v4f*)(src + 12);
    __bf16 e[16];
    #pragma unroll
    for (int j = 0; j < 4; ++j) {
      e[j]      = f2bf(w0[j]);
      e[j + 4]  = f2bf(w1[j]);
      e[j + 8]  = f2bf(w2[j]);
      e[j + 12] = f2bf(w3[j]);
    }
    const int pb = swz(blk) * 8;
    #pragma unroll
    for (int p = 0; p < 8; ++p) {
      v8bf v = { e[p], e[p+1], e[p+2], e[p+3], e[p+4], e[p+5], e[p+6], e[p+7] };
      *(v8bf*)&xcop[p][pb] = v;
    }
  }
  __syncthreads();

  // ---- Stage 2: WMMA correlation, lags 0..2047 (r[s] = r[T-s] symmetry) ----
  // wave = (pg, h): parity pair {2pg, 2pg+1}, t0 quarter h. One A fetch feeds two B's.
  // D[m][n] = sum_k x[t0+k+8m] * x[t0+k+120+128n+p]  ->  lag = p + 120 - 8m + 128n
  const int wave = tid >> 5, lane = tid & 31;
  const int pg = wave & 3, h = wave >> 2;
  const int p0 = 2 * pg, p1 = p0 + 1;
  const int lh = lane & 15, hh = lane >> 4;

  const int aL = lh + hh;             // A block offset (+= t0/8); chunks at +0, +2
  const int bL = 15 + 16 * lh + hh;   // B block offset (+= t0/8); chunks at +0, +2

  const __bf16* c0  = &xcop[0][0];
  const __bf16* cp0 = &xcop[p0][0];
  const __bf16* cp1 = &xcop[p1][0];

  v8f acc0 = {0.f,0.f,0.f,0.f,0.f,0.f,0.f,0.f};
  v8f acc1 = {0.f,0.f,0.f,0.f,0.f,0.f,0.f,0.f};

  #pragma unroll 4
  for (int it = 0; it < 32; ++it) {
    const int tb = h * 128 + it * 4;      // t0/8, t0 in [1024h, 1024h+1024) step 32
    const int sa0 = swz(tb + aL) * 8;
    const int sa1 = swz(tb + aL + 2) * 8;
    const int sb0 = swz(tb + bL) * 8;
    const int sb1 = swz(tb + bL + 2) * 8;
    v8bf a0  = *(const v8bf*)(c0 + sa0);
    v8bf a1  = *(const v8bf*)(c0 + sa1);
    v16bf A  = cat16(a0, a1);
    v8bf b00 = *(const v8bf*)(cp0 + sb0);
    v8bf b01 = *(const v8bf*)(cp0 + sb1);
    acc0 = __builtin_amdgcn_wmma_f32_16x16x32_bf16(
        false, A, false, cat16(b00, b01), (short)0, acc0, false, false);
    v8bf b10 = *(const v8bf*)(cp1 + sb0);
    v8bf b11 = *(const v8bf*)(cp1 + sb1);
    acc1 = __builtin_amdgcn_wmma_f32_16x16x32_bf16(
        false, A, false, cat16(b10, b11), (short)0, acc1, false, false);
  }

  // D layout: VGPR v holds M = v + 8*hh, N = lh. 4 t0-quarter waves merge via ds_add.
  #pragma unroll
  for (int v = 0; v < 8; ++v) {
    const int M = v + 8 * hh;
    const int lag0 = p0 + 120 - 8 * M + 128 * lh;   // in [0, 2047]
    atomicAdd(&rarr[ridx(lag0)],     acc0[v]);
    atomicAdd(&rarr[ridx(lag0 + 1)], acc1[v]);
  }

  // lag 2048 (self-mirror) via fp32 partial sums
  {
    float part = 0.0f;
    for (int t = tid; t < TT; t += NTHREADS)
      part += xrowext[t] * xrowext[t + 2048];
    atomicAdd(&rarr[ridx(2048)], part);
  }
  __syncthreads();

  // ---- Stage 3: top-3 over mirrored r; stable (lower index wins ties) like lax.top_k ----
  for (int k = 0; k < 3; ++k) {
    float bv = -3.402823466e+38f;
    int   bi = 0x7FFFFFFF;
    for (int s = tid; s < TT; s += NTHREADS) {
      const int ms = (s <= 2048) ? s : (TT - s);
      const float v = rarr[ridx(ms)];
      bool skip = false;
      for (int q = 0; q < k; ++q) skip = skip || (lagsSh[q] == s);
      if (!skip && (v > bv || (v == bv && s < bi))) { bv = v; bi = s; }
    }
    rv[tid] = bv; ri[tid] = bi;
    __syncthreads();
    for (int off = NTHREADS / 2; off > 0; off >>= 1) {
      if (tid < off) {
        const float ov = rv[tid + off]; const int oi = ri[tid + off];
        if (ov > rv[tid] || (ov == rv[tid] && oi < ri[tid])) { rv[tid] = ov; ri[tid] = oi; }
      }
      __syncthreads();
    }
    if (tid == 0) lagsSh[k] = ri[0];
    __syncthreads();
  }

  // ---- Stage 4: aggregate delayed copies, masked to valid (t - s >= 0), /3 ----
  const int s0 = lagsSh[0], s1 = lagsSh[1], s2 = lagsSh[2];
  const float inv3 = 1.0f / 3.0f;
  for (int t = tid; t < TT; t += NTHREADS) {
    float sum = 0.0f;
    if (t >= s0) sum += xrowext[t - s0];
    if (t >= s1) sum += xrowext[t - s1];
    if (t >= s2) sum += xrowext[t - s2];
    out[base + t] = sum * inv3;
  }
}

extern "C" void kernel_launch(void* const* d_in, const int* in_sizes, int n_in,
                              void* d_out, int out_size, void* d_ws, size_t ws_size,
                              hipStream_t stream) {
  const float* x = (const float*)d_in[0];
  float* out = (float*)d_out;
  const int nrows = in_sizes[0] / TT;   // 32*512 = 16384 rows of length 4096
  AutoCorrelation_gfx1250_kernel<<<nrows, NTHREADS, 0, stream>>>(x, out);
}
